// BlockCirculant_18279380812373
// MI455X (gfx1250) — matrix operand order repaired and checked by reading, not compiled, not run on hardware
//
#include <hip/hip_runtime.h>

// ---------------------------------------------------------------------------
// Block-circulant layer as a structured GEMM on CDNA5 WMMA (bf16x3 split).
//   y = (x * d) @ Bw + bias,  Bw[j*256+kk, i*256+nn] = W[i, j, (kk-nn) & 255]
// - LDS holds bf16 hi/lo planes in WMMA-fragment order: every fragment is
//   two ds_load_b128 per lane (no unpack VALU).
// - x tiles are double-pumped with GLOBAL_LOAD_ASYNC_TO_LDS_B128: the copy of
//   tile t+1 overlaps the WMMA phase of tile t (ASYNCcnt-tracked).
// ---------------------------------------------------------------------------

typedef __attribute__((ext_vector_type(16))) __bf16        v16bf;
typedef __attribute__((ext_vector_type(8)))  float         v8f;
typedef __attribute__((ext_vector_type(4)))  float         v4f;
typedef __attribute__((ext_vector_type(4)))  unsigned int  v4u;

#define NROWS   8192
#define DIN     4096
#define DOUT    4096
#define BLK_M   128
#define BLK_N   128
#define BLK_K   32
#define AROW    40    // ushorts per A-plane row (80B: 16B-aligned, conflict-free)
#define BROW    40    // ushorts per B-plane row
#define WROW    272   // u32 per W row: 256 + 16 wrap margin (gather reads c0..c0+15)

#define SEL_HI  0x07060302u  // v_perm: {b3,b2} of each src -> (u >> 16) pair
#define SEL_LO  0x05040100u  // v_perm: {b1,b0} of each src -> (u & 0xffff) pair

union FragBF {
    v16bf v;
    v4u   q[2];
};

__device__ __forceinline__ unsigned int prm(unsigned int a1, unsigned int a0,
                                            unsigned int sel) {
    return __builtin_amdgcn_perm(a1, a0, sel);
}

__device__ __forceinline__ void load_frag(FragBF& f, const unsigned short* p16) {
    const v4u* p = (const v4u*)p16;   // 16B-aligned by construction
    f.q[0] = p[0];
    f.q[1] = p[1];
}

__device__ __forceinline__ v8f wmma_bf16(const FragBF& a, const FragBF& b, v8f c) {
    return __builtin_amdgcn_wmma_f32_16x16x32_bf16(
        false, a.v, false, b.v, (short)0, c, false, false);
}

// Async-copy this thread's 4x16B slice of the x tile at column k0 into LDS.
__device__ __forceinline__ void async_x_tile(const float* __restrict__ x,
                                             int m0, int k0, int srow, int scol,
                                             unsigned int lds_base) {
#pragma unroll
    for (int p = 0; p < 4; ++p) {
        const int row = srow + p * 32;
        const unsigned int dst = lds_base + (unsigned int)((row * 32 + scol) * 4);
        const unsigned long long src =
            (unsigned long long)(x + (size_t)(m0 + row) * DIN + k0 + scol);
        asm volatile("global_load_async_to_lds_b128 %0, %1, off"
                     :: "v"(dst), "v"(src) : "memory");
    }
}

__global__ __launch_bounds__(256)
void bc_wmma_kernel(const float* __restrict__ x,
                    const float* __restrict__ W,      // [16][16][256] f32
                    const float* __restrict__ dsign,  // [4096] f32 (+-1)
                    const float* __restrict__ bias,   // [4096] f32
                    float* __restrict__ out)          // [8192][4096] f32
{
    __shared__ __align__(16) float          lds_x[BLK_M * BLK_K];  // raw x tile
    __shared__ __align__(16) unsigned short a_hi[BLK_M * AROW];
    __shared__ __align__(16) unsigned short a_lo[BLK_M * AROW];
    __shared__ __align__(16) unsigned short b_hi[BLK_N * BROW];
    __shared__ __align__(16) unsigned short b_lo[BLK_N * BROW];
    __shared__ unsigned int w_pk[16 * WROW];  // {hi16,lo16} packed W panel

    const int tid     = threadIdx.x;
    const int lane    = tid & 31;
    const int hi_lane = (lane >> 4) & 1;
    const int l15     = lane & 15;
    const int wave    = tid >> 5;      // 0..7
    const int wave_m  = wave >> 2;     // 0..1 -> 64-row strip
    const int wave_n  = wave & 3;      // 0..3 -> 32-col strip

    const int n0      = blockIdx.x * BLK_N;
    const int m0      = blockIdx.y * BLK_M;
    const int iblk    = n0 >> 8;       // circulant output block (128 | 256)
    const int nn_base = n0 & 255;

    // LDS byte address of lds_x (flat address low 32 bits == LDS offset)
    const unsigned int ldsx_base = (unsigned int)(size_t)&lds_x[0];

    // A staging: thread owns 4 consecutive K cols; K-group order [0,1,2,3] (contiguous)
    const int scol = (tid & 7) * 4;
    const int srow = tid >> 3;
    const int g    = scol >> 3;
    const int sg   = g;
    const int ap   = sg * 8 + (scol & 7);      // ushort position within A row

    // B staging: thread owns 16 consecutive kk of one column
    const int bt_n = tid >> 1;                  // 0..127
    const int kk0  = (tid & 1) * 16;

    // ---- kick off async copy of the first x tile ----
    async_x_tile(x, m0, 0, srow, scol, ldsx_base);

    // ---- preload + split W[iblk,:,:] into packed {hi,lo} with wrap margin ----
    for (int idx = tid; idx < 16 * WROW; idx += 256) {
        const int jj = idx / WROW;
        const int m  = idx - jj * WROW;
        const float wv = W[iblk * 4096 + jj * 256 + (m & 255)];
        const unsigned int u  = __float_as_uint(wv);
        const unsigned int fh = u & 0xffff0000u;
        const float        fl = wv - __uint_as_float(fh);
        w_pk[idx] = fh | (__float_as_uint(fl) >> 16);
    }

    // fragment base pointers (k-invariant; fm/fn offsets become ds immediates)
    const unsigned short* abase_h = a_hi + (wave_m * 64 + l15) * AROW + hi_lane * 16;
    const unsigned short* abase_l = a_lo + (wave_m * 64 + l15) * AROW + hi_lane * 16;
    const unsigned short* bbase_h = b_hi + (wave_n * 32 + l15) * BROW + hi_lane * 16;
    const unsigned short* bbase_l = b_lo + (wave_n * 32 + l15) * BROW + hi_lane * 16;

    v8f acc[4][2];
    const v8f vzero = {0.f, 0.f, 0.f, 0.f, 0.f, 0.f, 0.f, 0.f};
#pragma unroll
    for (int fm = 0; fm < 4; ++fm)
#pragma unroll
        for (int fn = 0; fn < 2; ++fn)
            acc[fm][fn] = vzero;

    for (int k0 = 0; k0 < DIN; k0 += BLK_K) {
        // this wave's slice of the x tile is ready once its ASYNCcnt drains
        asm volatile("s_wait_asynccnt 0x0" ::: "memory");
        __syncthreads();  // planes free (prev iter's fragment loads complete)

        // ---- stage A tile: split (x*d) from LDS into bf16 hi/lo planes ----
        const v4f dv = *(const v4f*)(dsign + k0 + scol);
#pragma unroll
        for (int p = 0; p < 4; ++p) {
            const int row = srow + p * 32;
            const v4f xv = *(const v4f*)&lds_x[row * 32 + scol];
            const float f0 = xv.x * dv.x, f1 = xv.y * dv.y;
            const float f2 = xv.z * dv.z, f3 = xv.w * dv.w;
            const unsigned int u0 = __float_as_uint(f0), u1 = __float_as_uint(f1);
            const unsigned int u2 = __float_as_uint(f2), u3 = __float_as_uint(f3);
            const unsigned int l0 = __float_as_uint(f0 - __uint_as_float(u0 & 0xffff0000u));
            const unsigned int l1 = __float_as_uint(f1 - __uint_as_float(u1 & 0xffff0000u));
            const unsigned int l2 = __float_as_uint(f2 - __uint_as_float(u2 & 0xffff0000u));
            const unsigned int l3 = __float_as_uint(f3 - __uint_as_float(u3 & 0xffff0000u));
            unsigned short* arow_h = a_hi + row * AROW + ap;
            unsigned short* arow_l = a_lo + row * AROW + ap;
            *(unsigned int*)(arow_h)     = prm(u1, u0, SEL_HI);
            *(unsigned int*)(arow_h + 2) = prm(u3, u2, SEL_HI);
            *(unsigned int*)(arow_l)     = prm(l1, l0, SEL_HI);
            *(unsigned int*)(arow_l + 2) = prm(l3, l2, SEL_HI);
        }

        // ---- stage B tile: circulant gather from W panel, fragment order ----
        {
            const int j = k0 >> 8;
            const unsigned int* wr = w_pk + j * WROW;
            const int c0 = ((k0 & 255) + kk0 - nn_base - bt_n) & 255;  // +15 < 272
            unsigned int q[16];
#pragma unroll
            for (int i = 0; i < 16; ++i)
                q[i] = wr[c0 + i];
            v4u h0, h1, ll0, ll1;
            h0.x  = prm(q[1],  q[0],  SEL_HI);  h0.y  = prm(q[3],  q[2],  SEL_HI);
            h0.z  = prm(q[5],  q[4],  SEL_HI);  h0.w  = prm(q[7],  q[6],  SEL_HI);
            h1.x  = prm(q[9],  q[8],  SEL_HI);  h1.y  = prm(q[11], q[10], SEL_HI);
            h1.z  = prm(q[13], q[12], SEL_HI);  h1.w  = prm(q[15], q[14], SEL_HI);
            ll0.x = prm(q[1],  q[0],  SEL_LO);  ll0.y = prm(q[3],  q[2],  SEL_LO);
            ll0.z = prm(q[5],  q[4],  SEL_LO);  ll0.w = prm(q[7],  q[6],  SEL_LO);
            ll1.x = prm(q[9],  q[8],  SEL_LO);  ll1.y = prm(q[11], q[10], SEL_LO);
            ll1.z = prm(q[13], q[12], SEL_LO);  ll1.w = prm(q[15], q[14], SEL_LO);
            *(v4u*)(b_hi + bt_n * BROW + kk0)     = h0;
            *(v4u*)(b_hi + bt_n * BROW + kk0 + 8) = h1;
            *(v4u*)(b_lo + bt_n * BROW + kk0)     = ll0;
            *(v4u*)(b_lo + bt_n * BROW + kk0 + 8) = ll1;
        }

        // lds_x fully consumed by this wave -> safe to overwrite with next tile
        // (async LDS writes are unordered vs DS ops, so drain DScnt first)
        if (k0 + BLK_K < DIN) {
            asm volatile("s_wait_dscnt 0x0" ::: "memory");
            async_x_tile(x, m0, k0 + BLK_K, srow, scol, ldsx_base);
        }

        __syncthreads();

        // ---- fragments: 2x ds_load_b128 each, then bf16x3 WMMA ----
        FragBF bh[2], bl[2];
#pragma unroll
        for (int fn = 0; fn < 2; ++fn) {
            load_frag(bh[fn], bbase_h + fn * 16 * BROW);
            load_frag(bl[fn], bbase_l + fn * 16 * BROW);
        }
#pragma unroll
        for (int fm = 0; fm < 4; ++fm) {
            FragBF ah, al;
            load_frag(ah, abase_h + fm * 16 * AROW);
            load_frag(al, abase_l + fm * 16 * AROW);
#pragma unroll
            for (int fn = 0; fn < 2; ++fn) {
                v8f c = acc[fm][fn];
                c = wmma_bf16(ah, bh[fn], c);
                c = wmma_bf16(ah, bl[fn], c);
                c = wmma_bf16(al, bh[fn], c);
                acc[fm][fn] = c;
            }
        }
    }

    // ---- epilogue: C/D layout lane=N, VGPR r -> M=r (+8 for lanes 16-31) ----
#pragma unroll
    for (int fm = 0; fm < 4; ++fm)
#pragma unroll
        for (int fn = 0; fn < 2; ++fn) {
            const int n_g    = n0 + wave_n * 32 + fn * 16 + l15;
            const float bv   = bias[n_g];
            const int m_base = m0 + wave_m * 64 + fm * 16 + 8 * hi_lane;
#pragma unroll
            for (int r = 0; r < 8; ++r)
                out[(size_t)(m_base + r) * DOUT + n_g] = acc[fm][fn][r] + bv;
        }
}

extern "C" void kernel_launch(void* const* d_in, const int* in_sizes, int n_in,
                              void* d_out, int out_size, void* d_ws, size_t ws_size,
                              hipStream_t stream) {
    (void)in_sizes; (void)n_in; (void)out_size; (void)d_ws; (void)ws_size;
    const float* x    = (const float*)d_in[0];
    const float* W    = (const float*)d_in[1];
    const float* dber = (const float*)d_in[2];
    const float* bias = (const float*)d_in[3];
    float* out = (float*)d_out;

    dim3 grid(DOUT / BLK_N, NROWS / BLK_M);  // (32, 64); x-major -> A-panel L2 reuse
    bc_wmma_kernel<<<grid, dim3(256), 0, stream>>>(x, W, dber, bias, out);
}
